// AdaptiveMultiHeadAttention_76373108457731
// MI455X (gfx1250) — matrix operand order repaired and checked by reading, not compiled, not run on hardware
//
#include <hip/hip_runtime.h>
#include <hip/hip_bf16.h>

// Problem constants (from reference): B=4, S=2048, D=512, H=8, dk=64
#define BATCH   4
#define S_LEN   2048
#define DMODEL  512
#define NHEAD   8
#define DK      64
#define LN_EPS  1e-5f

typedef __bf16 bf16;
typedef __attribute__((ext_vector_type(16))) __bf16 v16bf;
typedef __attribute__((ext_vector_type(8)))  float  v8f;

static __device__ __forceinline__ v8f wmma_bf16(v16bf a, v16bf b, v8f c) {
  // D = A(16x32 bf16) x B(32x16 bf16) + C(16x16 f32)
  return __builtin_amdgcn_wmma_f32_16x16x32_bf16(
      /*neg_a=*/false, a, /*neg_b=*/false, b,
      /*c_mod=*/(short)0, c, /*reuse_a=*/false, /*reuse_b=*/false);
}

static __device__ __forceinline__ v8f vzero8() {
  v8f z;
#pragma unroll
  for (int i = 0; i < 8; ++i) z[i] = 0.0f;
  return z;
}

// ---- CDNA5 async global->LDS copy (ASYNCcnt-tracked) ------------------------
// LDS byte address = low 32 bits of the flat pointer (ISA 10.2: LDS aperture
// addresses are truncated to [31:0] and used directly as the LDS offset).
static __device__ __forceinline__ void async_ld16(void* lds_dst, const void* gsrc) {
  asm volatile("global_load_async_to_lds_b128 %0, %1, off"
               :
               : "v"((unsigned int)(uintptr_t)lds_dst), "v"(gsrc)
               : "memory");
}
static __device__ __forceinline__ void wait_async0() {
  asm volatile("s_wait_asynccnt 0x0" ::: "memory");
}

// ---- Fragment loaders (wave32 layouts per CDNA5 ISA 7.12.2) -----------------
// A fragment 16(M) x 32(K), source row-major (row stride ld, bf16):
//   lane holds row (lane&15); K elements at kb+{0..7} and kb+16+{0..7}, kb=8*(lane>>4)
static __device__ __forceinline__ v16bf fragA_bf16(const bf16* __restrict__ base, int ld) {
  const int lane = threadIdx.x & 31;
  const bf16* p = base + (size_t)(lane & 15) * ld + ((lane >> 4) << 3);
  union { uint4 u; bf16 h[8]; } lo, hi;
  lo.u = *(const uint4*)(p);
  hi.u = *(const uint4*)(p + 16);
  v16bf f;
#pragma unroll
  for (int e = 0; e < 8; ++e) { f[e] = lo.h[e]; f[8 + e] = hi.h[e]; }
  return f;
}

// Same A layout, but converting from fp32 source on the fly.
static __device__ __forceinline__ v16bf fragA_f32(const float* __restrict__ base, int ld) {
  const int lane = threadIdx.x & 31;
  const float* p = base + (size_t)(lane & 15) * ld + ((lane >> 4) << 3);
  v16bf f;
#pragma unroll
  for (int e = 0; e < 8; ++e) {
    f[e]     = (bf16)p[e];
    f[8 + e] = (bf16)p[16 + e];
  }
  return f;
}

// B fragment 32(K) x 16(N). Source is the (N rows, K cols) row-major view of
// B^T (i.e. we pass W with W[n][k] when B[k][n]=W[n][k]). Lane holds column
// (lane&15); K elements at kb+{0..7} and kb+16+{0..7}, kb=8*(lane>>4).
static __device__ __forceinline__ v16bf fragB_bf16(const bf16* __restrict__ base, int ld) {
  const int lane = threadIdx.x & 31;
  const bf16* p = base + (size_t)(lane & 15) * ld + ((lane >> 4) << 3);
  union { uint4 u; bf16 h[8]; } lo, hi;
  lo.u = *(const uint4*)(p);
  hi.u = *(const uint4*)(p + 16);
  v16bf f;
#pragma unroll
  for (int e = 0; e < 8; ++e) { f[e] = lo.h[e]; f[8 + e] = hi.h[e]; }
  return f;
}

// ---- small prep kernels -----------------------------------------------------
__global__ __launch_bounds__(256) void cvt_f32_bf16_kernel(
    const float* __restrict__ src, bf16* __restrict__ dst, int n) {
  int i = blockIdx.x * 256 + threadIdx.x;
  if (i < n) dst[i] = (bf16)src[i];
}

// ew[b][s] = exp(We[s] * entropy[b][s])   (entropy is (B,S,1) flat)
__global__ __launch_bounds__(256) void ew_kernel(
    const float* __restrict__ We, const float* __restrict__ entropy,
    float* __restrict__ ew) {
  int i = blockIdx.x * 256 + threadIdx.x;   // i < B*S
  int s = i & (S_LEN - 1);
  ew[i] = __expf(We[s] * entropy[i]);
}

// ---- QKV projection GEMM: y[m,n] = sum_k X[m,k]*W[n,k] + bias[n] ------------
// M = B*S = 8192, N = D = 512, K = D = 512.
// Block = 256 thr (8 waves, 4x2), block tile 128x64, wave tile 32x32.
// transposed==0 : out (B,H,S,dk) bf16     (Q, K projections)
// transposed==1 : out (B,H,dk,S) bf16     (V projection, pre-transposed)
__global__ __launch_bounds__(256) void qkv_proj_kernel(
    const float* __restrict__ X, const bf16* __restrict__ Wbf,
    const float* __restrict__ bias, bf16* __restrict__ outp, int transposed) {
  const int wave = threadIdx.x >> 5;
  const int m0 = blockIdx.x * 128 + (wave >> 1) * 32;
  const int n0 = blockIdx.y * 64 + (wave & 1) * 32;

  v8f acc[2][2];
#pragma unroll
  for (int i = 0; i < 2; ++i)
#pragma unroll
    for (int j = 0; j < 2; ++j) acc[i][j] = vzero8();

  for (int k0 = 0; k0 < DMODEL; k0 += 32) {
    v16bf a0 = fragA_f32(X + (size_t)m0 * DMODEL + k0, DMODEL);
    v16bf a1 = fragA_f32(X + (size_t)(m0 + 16) * DMODEL + k0, DMODEL);
    v16bf b0 = fragB_bf16(Wbf + (size_t)n0 * DMODEL + k0, DMODEL);
    v16bf b1 = fragB_bf16(Wbf + (size_t)(n0 + 16) * DMODEL + k0, DMODEL);
    acc[0][0] = wmma_bf16(a0, b0, acc[0][0]);
    acc[0][1] = wmma_bf16(a0, b1, acc[0][1]);
    acc[1][0] = wmma_bf16(a1, b0, acc[1][0]);
    acc[1][1] = wmma_bf16(a1, b1, acc[1][1]);
  }

  const int lane = threadIdx.x & 31;
  const int nc = lane & 15, hi = lane >> 4;
#pragma unroll
  for (int mi = 0; mi < 2; ++mi) {
#pragma unroll
    for (int ni = 0; ni < 2; ++ni) {
      const int n = n0 + ni * 16 + nc;
      const float bv = bias[n];
      const int h = n >> 6, d = n & (DK - 1);
#pragma unroll
      for (int j = 0; j < 8; ++j) {
        const int m = m0 + mi * 16 + j + 8 * hi;      // C layout: row j+8*hi
        const int bb = m >> 11, s = m & (S_LEN - 1);
        const float y = acc[mi][ni][j] + bv;
        if (transposed)
          outp[(((size_t)bb * NHEAD + h) * DK + d) * S_LEN + s] = (bf16)y;
        else
          outp[(((size_t)bb * NHEAD + h) * S_LEN + s) * DK + d] = (bf16)y;
      }
    }
  }
}

// ---- Flash attention --------------------------------------------------------
// One wave per 16-query tile; 8 waves/block share one (b,h) and cooperatively
// double-buffer the K (32x64) and V^T (64x32) chunks into LDS with CDNA5
// async global->LDS copies (ASYNCcnt + barrier pipeline). Fragments are then
// read from LDS (ds_load_b128).
// scores^T = K x Q^T via WMMA (lane owns query column q=lane&15); online
// softmax with scale*ew per key; P tile feeds P x V WMMA as an A-fragment
// with zero cross-lane data movement.
__global__ __launch_bounds__(256) void flash_attn_kernel(
    const bf16* __restrict__ Qp,   // (B,H,S,dk)
    const bf16* __restrict__ Kp,   // (B,H,S,dk)
    const bf16* __restrict__ Vt,   // (B,H,dk,S)
    const float* __restrict__ ew,  // (B,S)
    bf16* __restrict__ attn_out)   // (B,S,D) bf16
{
  __shared__ __align__(16) bf16 kbuf[2][32 * DK];   // 2 x 4 KB, rows kv, ld=64
  __shared__ __align__(16) bf16 vbuf[2][DK * 32];   // 2 x 4 KB, rows d,  ld=32

  const int wave = threadIdx.x >> 5;
  const int tile = blockIdx.x * 8 + wave;            // 0 .. B*H*(S/16)-1
  const int qt = tile & (S_LEN / 16 - 1);
  const int h  = (tile >> 7) & (NHEAD - 1);          // uniform within block
  const int b  = tile >> 10;                         // uniform within block
  const int lane = threadIdx.x & 31;
  const int qc = lane & 15, hi = lane >> 4;
  const int tid = threadIdx.x;

  const bf16* qbase = Qp + (((size_t)b * NHEAD + h) * S_LEN + (size_t)qt * 16) * DK;
  const bf16* kbase = Kp + (((size_t)b * NHEAD + h) * S_LEN) * DK;
  const bf16* vbase = Vt + (((size_t)b * NHEAD + h) * DK) * S_LEN;
  const float* ewb  = ew + (size_t)b * S_LEN;

  // Per-thread 16B slices of each chunk:
  //   K chunk (4 KB contiguous): thread tid copies bytes [tid*16, tid*16+16)
  //   V chunk: row d = tid>>2 (stride S_LEN in global, 32 in LDS), seg tid&3
  const int vrow = tid >> 2, vseg = (tid & 3) * 8;

  // Prologue: issue async copies for chunk 0 into buffer 0.
  async_ld16(&kbuf[0][tid * 8], kbase + (size_t)tid * 8);
  async_ld16(&vbuf[0][vrow * 32 + vseg], vbase + (size_t)vrow * S_LEN + vseg);

  // Q as B-fragments (N=query col, K=d): two d-chunks of 32 (direct global)
  v16bf qf0 = fragB_bf16(qbase + 0,  DK);
  v16bf qf1 = fragB_bf16(qbase + 32, DK);

  v8f oacc[4];
#pragma unroll
  for (int t = 0; t < 4; ++t) oacc[t] = vzero8();
  float m_run = -1e30f, l_run = 0.0f;
  const float scale = 8.0f;                          // sqrt(dk); ref MULTIPLIES

  for (int kv0 = 0; kv0 < S_LEN; kv0 += 32) {
    const int buf = (kv0 >> 5) & 1;
    // Our own async copies for `buf` complete, and (via the barrier) so has
    // every other wave's copy and every wave's reads of buffer buf^1.
    wait_async0();
    __syncthreads();

    if (kv0 + 32 < S_LEN) {                          // uniform branch
      const int nb = buf ^ 1;
      async_ld16(&kbuf[nb][tid * 8],
                 kbase + (size_t)(kv0 + 32) * DK + (size_t)tid * 8);
      async_ld16(&vbuf[nb][vrow * 32 + vseg],
                 vbase + (size_t)vrow * S_LEN + (kv0 + 32) + vseg);
    }

    // scores^T tiles from LDS: rows kv (two 16-row tiles), cols q
    const bf16* kp = &kbuf[buf][0];
    v16bf kf0a = fragA_bf16(kp,                DK);  // kv 0..15 , d 0..31
    v16bf kf0b = fragA_bf16(kp + 32,           DK);  // kv 0..15 , d 32..63
    v16bf kf1a = fragA_bf16(kp + 16 * DK,      DK);  // kv 16..31, d 0..31
    v16bf kf1b = fragA_bf16(kp + 16 * DK + 32, DK);
    v8f st0 = vzero8(), st1 = vzero8();
    st0 = wmma_bf16(kf0a, qf0, st0); st0 = wmma_bf16(kf0b, qf1, st0);
    st1 = wmma_bf16(kf1a, qf0, st1); st1 = wmma_bf16(kf1b, qf1, st1);

    // lane covers kv = kv0+8*hi+j (tile0) and kv0+16+8*hi+j (tile1), col q=qc
    float sv[16];
    float cmax = -1e30f;
#pragma unroll
    for (int j = 0; j < 8; ++j) {
      const float e0 = ewb[kv0 + 8 * hi + j];
      const float e1 = ewb[kv0 + 16 + 8 * hi + j];
      sv[j]     = st0[j] * scale * e0;
      sv[8 + j] = st1[j] * scale * e1;
      cmax = fmaxf(cmax, fmaxf(sv[j], sv[8 + j]));
    }
    cmax = fmaxf(cmax, __shfl_xor(cmax, 16, 32));    // both halves of column q
    const float m_new = fmaxf(m_run, cmax);
    const float alpha = __expf(m_run - m_new);
    m_run = m_new;
    l_run *= alpha;

    // P as A-fragment of (P x V): row q=lane&15, K striping kb=8*hi matches
    // exactly the values this lane already holds. No shuffles needed.
    v16bf pf;
    float psum = 0.0f;
#pragma unroll
    for (int j = 0; j < 8; ++j) {
      const float p0 = __expf(sv[j] - m_new);
      const float p1 = __expf(sv[8 + j] - m_new);
      psum += p0 + p1;
      pf[j]     = (bf16)p0;
      pf[8 + j] = (bf16)p1;
    }
    l_run += psum;

    // Rescale accumulators: out C-layout row q = j + 8*hi, alpha lives in
    // lane with (lane&15)==q -> broadcast via shfl from lanes 0..15.
#pragma unroll
    for (int j = 0; j < 8; ++j) {
      const float al = __shfl(alpha, 8 * hi + j, 32);
#pragma unroll
      for (int t = 0; t < 4; ++t) oacc[t][j] *= al;
    }

    // out[q, d] += P(16x32) x V(32x16) per d-tile; B-frag from LDS V (ld=32)
#pragma unroll
    for (int t = 0; t < 4; ++t) {
      v16bf vf = fragB_bf16(&vbuf[buf][(t * 16) * 32], 32);
      oacc[t] = wmma_bf16(pf, vf, oacc[t]);
    }
  }

  const float l_tot = l_run + __shfl_xor(l_run, 16, 32);
  const float inv = 1.0f / l_tot;

  // Store: out row q = j+8*hi (global s = qt*16+q), col d = t*16 + qc
#pragma unroll
  for (int j = 0; j < 8; ++j) {
    const int q = j + 8 * hi;
    const float invq = __shfl(inv, q, 32);
    bf16* orow = attn_out + ((size_t)b * S_LEN + (size_t)qt * 16 + q) * DMODEL
                 + h * DK + qc;
#pragma unroll
    for (int t = 0; t < 4; ++t) orow[t * 16] = (bf16)(oacc[t][j] * invq);
  }
}

// ---- FC projection + bias + residual (fp32 out) -----------------------------
__global__ __launch_bounds__(256) void fc_gemm_kernel(
    const bf16* __restrict__ A, const bf16* __restrict__ Wbf,
    const float* __restrict__ bias, const float* __restrict__ res,
    float* __restrict__ outp) {
  const int wave = threadIdx.x >> 5;
  const int m0 = blockIdx.x * 128 + (wave >> 1) * 32;
  const int n0 = blockIdx.y * 64 + (wave & 1) * 32;

  v8f acc[2][2];
#pragma unroll
  for (int i = 0; i < 2; ++i)
#pragma unroll
    for (int j = 0; j < 2; ++j) acc[i][j] = vzero8();

  for (int k0 = 0; k0 < DMODEL; k0 += 32) {
    v16bf a0 = fragA_bf16(A + (size_t)m0 * DMODEL + k0, DMODEL);
    v16bf a1 = fragA_bf16(A + (size_t)(m0 + 16) * DMODEL + k0, DMODEL);
    v16bf b0 = fragB_bf16(Wbf + (size_t)n0 * DMODEL + k0, DMODEL);
    v16bf b1 = fragB_bf16(Wbf + (size_t)(n0 + 16) * DMODEL + k0, DMODEL);
    acc[0][0] = wmma_bf16(a0, b0, acc[0][0]);
    acc[0][1] = wmma_bf16(a0, b1, acc[0][1]);
    acc[1][0] = wmma_bf16(a1, b0, acc[1][0]);
    acc[1][1] = wmma_bf16(a1, b1, acc[1][1]);
  }

  const int lane = threadIdx.x & 31;
  const int nc = lane & 15, hi = lane >> 4;
#pragma unroll
  for (int mi = 0; mi < 2; ++mi) {
#pragma unroll
    for (int ni = 0; ni < 2; ++ni) {
      const int n = n0 + ni * 16 + nc;
      const float bv = bias[n];
#pragma unroll
      for (int j = 0; j < 8; ++j) {
        const int m = m0 + mi * 16 + j + 8 * hi;
        outp[(size_t)m * DMODEL + n] =
            acc[mi][ni][j] + bv + res[(size_t)m * DMODEL + n];
      }
    }
  }
}

// ---- LayerNorm over D=512, in place on d_out --------------------------------
__global__ __launch_bounds__(256) void layernorm_kernel(
    float* __restrict__ out, const float* __restrict__ gamma,
    const float* __restrict__ beta) {
  float* p = out + (size_t)blockIdx.x * DMODEL;
  const int t = threadIdx.x;
  const float x0 = p[t], x1 = p[t + 256];
  float s = x0 + x1, ss = x0 * x0 + x1 * x1;
#pragma unroll
  for (int off = 16; off >= 1; off >>= 1) {
    s  += __shfl_xor(s, off, 32);
    ss += __shfl_xor(ss, off, 32);
  }
  __shared__ float sums[8], sqs[8];
  const int w = t >> 5;
  if ((t & 31) == 0) { sums[w] = s; sqs[w] = ss; }
  __syncthreads();
  float S = 0.0f, SS = 0.0f;
#pragma unroll
  for (int i = 0; i < 8; ++i) { S += sums[i]; SS += sqs[i]; }
  const float mu = S * (1.0f / DMODEL);
  const float var = SS * (1.0f / DMODEL) - mu * mu;
  const float rstd = rsqrtf(var + LN_EPS);
  p[t]       = (x0 - mu) * rstd * gamma[t]       + beta[t];
  p[t + 256] = (x1 - mu) * rstd * gamma[t + 256] + beta[t + 256];
}

// ---- host launch ------------------------------------------------------------
extern "C" void kernel_launch(void* const* d_in, const int* in_sizes, int n_in,
                              void* d_out, int out_size, void* d_ws, size_t ws_size,
                              hipStream_t stream) {
  const float* Q       = (const float*)d_in[0];
  const float* K       = (const float*)d_in[1];
  const float* V       = (const float*)d_in[2];
  const float* entropy = (const float*)d_in[3];
  const float* Wq      = (const float*)d_in[4];
  const float* bq      = (const float*)d_in[5];
  const float* Wk      = (const float*)d_in[6];
  const float* bk      = (const float*)d_in[7];
  const float* Wv      = (const float*)d_in[8];
  const float* bv      = (const float*)d_in[9];
  const float* Wfc     = (const float*)d_in[10];
  const float* bfc     = (const float*)d_in[11];
  const float* We      = (const float*)d_in[12];
  const float* gamma   = (const float*)d_in[13];
  const float* beta    = (const float*)d_in[14];
  float* out = (float*)d_out;

  const size_t WBYTES = (size_t)DMODEL * DMODEL * sizeof(bf16);        // 512 KB
  const size_t PBYTES = (size_t)BATCH * S_LEN * DMODEL * sizeof(bf16); // 8 MB
  char* ws = (char*)d_ws;
  bf16* wq_bf  = (bf16*)ws;                 ws += WBYTES;
  bf16* wk_bf  = (bf16*)ws;                 ws += WBYTES;
  bf16* wv_bf  = (bf16*)ws;                 ws += WBYTES;
  bf16* wfc_bf = (bf16*)ws;                 ws += WBYTES;
  bf16* Qp     = (bf16*)ws;                 ws += PBYTES;  // (B,H,S,dk)
  bf16* Kp     = (bf16*)ws;                 ws += PBYTES;  // (B,H,S,dk)
  bf16* Vt     = (bf16*)ws;                 ws += PBYTES;  // (B,H,dk,S)
  bf16* Abf    = (bf16*)ws;                 ws += PBYTES;  // attn out (B,S,D)
  float* ewbuf = (float*)ws;                ws += (size_t)BATCH * S_LEN * sizeof(float);

  const int WN = DMODEL * DMODEL;
  cvt_f32_bf16_kernel<<<(WN + 255) / 256, 256, 0, stream>>>(Wq,  wq_bf,  WN);
  cvt_f32_bf16_kernel<<<(WN + 255) / 256, 256, 0, stream>>>(Wk,  wk_bf,  WN);
  cvt_f32_bf16_kernel<<<(WN + 255) / 256, 256, 0, stream>>>(Wv,  wv_bf,  WN);
  cvt_f32_bf16_kernel<<<(WN + 255) / 256, 256, 0, stream>>>(Wfc, wfc_bf, WN);
  ew_kernel<<<(BATCH * S_LEN) / 256, 256, 0, stream>>>(We, entropy, ewbuf);

  dim3 gemm_grid(BATCH * S_LEN / 128, DMODEL / 64);
  qkv_proj_kernel<<<gemm_grid, 256, 0, stream>>>(Q, wq_bf, bq, Qp, 0);
  qkv_proj_kernel<<<gemm_grid, 256, 0, stream>>>(K, wk_bf, bk, Kp, 0);
  qkv_proj_kernel<<<gemm_grid, 256, 0, stream>>>(V, wv_bf, bv, Vt, 1);

  const int n_tiles = BATCH * NHEAD * (S_LEN / 16);   // 4096, 8 waves/block
  flash_attn_kernel<<<n_tiles / 8, 256, 0, stream>>>(Qp, Kp, Vt, ewbuf, Abf);

  fc_gemm_kernel<<<gemm_grid, 256, 0, stream>>>(Abf, wfc_bf, bfc, Q, out);
  layernorm_kernel<<<BATCH * S_LEN, 256, 0, stream>>>(out, gamma, beta);
}
